// DAAAv4_24481313587851
// MI455X (gfx1250) — matrix operand order repaired and checked by reading, not compiled, not run on hardware
//
#include <hip/hip_runtime.h>

typedef __attribute__((ext_vector_type(16))) __bf16 v16bf;
typedef __attribute__((ext_vector_type(8)))  float  v8f;
typedef __attribute__((ext_vector_type(4)))  float  v4f;

#define F 128  // F_IN == HID == 128

// ---------------- elementwise / utility kernels ----------------

__global__ void k_zero(float* __restrict__ p, long n) {
  long i = (long)blockIdx.x * blockDim.x + threadIdx.x;
  long s = (long)gridDim.x * blockDim.x;
  for (; i < n; i += s) p[i] = 0.f;
}

__global__ void k_xw(const float* __restrict__ x, const float* __restrict__ fi,
                     float* __restrict__ xw, long total) {
  long i = (long)blockIdx.x * blockDim.x + threadIdx.x;
  long s = (long)gridDim.x * blockDim.x;
  for (; i < total; i += s) {
    int f = (int)(i & (F - 1));
    float sg = 1.f / (1.f + __expf(-fi[f]));
    xw[i] = x[i] * sg;
  }
}

__global__ void k_deg(const int* __restrict__ row, const int* __restrict__ col,
                      float* __restrict__ degr, float* __restrict__ degc, int E) {
  int i = blockIdx.x * blockDim.x + threadIdx.x;
  int s = gridDim.x * blockDim.x;
  for (; i < E; i += s) {
    atomicAdd(&degr[row[i]], 1.f);
    atomicAdd(&degc[col[i]], 1.f);
  }
}

__global__ void k_dinv(const float* __restrict__ degc, float* __restrict__ dinv, int N) {
  int i = blockIdx.x * blockDim.x + threadIdx.x;
  int s = gridDim.x * blockDim.x;
  for (; i < N; i += s) dinv[i] = rsqrtf(degc[i] + 1.f);  // deg_sl = deg_col + 1 >= 1
}

// out[d] += src[s] * (dinv ? dinv[s]*dinv[d] : 1)   -- one edge per block iter, 128 feats
__global__ void k_scatter(const float* __restrict__ src, const int* __restrict__ sidx,
                          const int* __restrict__ didx, const float* __restrict__ dinv,
                          float* __restrict__ out, int E) {
  int f = threadIdx.x;
  for (int e = blockIdx.x; e < E; e += gridDim.x) {
    int s = sidx[e], d = didx[e];
    float scale = 1.f;
    if (dinv) scale = dinv[s] * dinv[d];
    atomicAdd(&out[(long)d * F + f], src[(long)s * F + f] * scale);
  }
}

__global__ void k_rowdiv(float* __restrict__ a, const float* __restrict__ deg, long total) {
  long i = (long)blockIdx.x * blockDim.x + threadIdx.x;
  long s = (long)gridDim.x * blockDim.x;
  for (; i < total; i += s) {
    int n = (int)(i >> 7);
    a[i] /= fmaxf(deg[n], 1.f);
  }
}

// GCN epilogue: out = maybe_relu(agg + hw * dinv^2 + b)
__global__ void k_gcn_epi(const float* __restrict__ agg, const float* __restrict__ hw,
                          const float* __restrict__ dinv, const float* __restrict__ b,
                          float* __restrict__ out, long total, int relu) {
  long i = (long)blockIdx.x * blockDim.x + threadIdx.x;
  long s = (long)gridDim.x * blockDim.x;
  for (; i < total; i += s) {
    int n = (int)(i >> 7), f = (int)(i & (F - 1));
    float di = dinv[n];
    float v = agg[i] + hw[i] * di * di + b[f];
    out[i] = relu ? fmaxf(v, 0.f) : v;
  }
}

// ---------------- per-node stats: finalize mean_nbr, dilution delta, gate MLP ----------------

__global__ void k_stats(const float* __restrict__ xw, float* __restrict__ mean_acc,
                        const float* __restrict__ degr,
                        const float* __restrict__ gw1, const float* __restrict__ gb1,
                        const float* __restrict__ gw2, const float* __restrict__ gb2,
                        float* __restrict__ gate, int N) {
  int n = (blockIdx.x * blockDim.x + threadIdx.x) >> 5;
  int lane = threadIdx.x & 31;
  if (n >= N) return;  // wave-uniform
  float deg = degr[n];
  float degc = fmaxf(deg, 1.f);
  float nx = 0.f, nm = 0.f, dot = 0.f;
#pragma unroll
  for (int i = 0; i < 4; ++i) {
    long idx = (long)n * F + lane + 32 * i;
    float xv = xw[idx];
    float mv = mean_acc[idx] / degc;
    mean_acc[idx] = mv;  // finalize mean_nbr == h_hop1
    nx += xv * xv; nm += mv * mv; dot += xv * mv;
  }
#pragma unroll
  for (int off = 16; off; off >>= 1) {
    nx  += __shfl_xor(nx, off, 32);
    nm  += __shfl_xor(nm, off, 32);
    dot += __shfl_xor(dot, off, 32);
  }
  float sim = dot / (fmaxf(sqrtf(nx), 1e-12f) * fmaxf(sqrtf(nm), 1e-12f));
  if (!(deg > 0.f)) sim = 1.f;
  float delta = 1.f / (1.f + __expf(-(deg * (1.f - sim) / 10.f - 0.5f)));
  // gate: hidden j == lane (32 hidden units)
  float gh = fmaxf(delta * gw1[lane] + gb1[lane], 0.f);
  float l0 = gh * gw2[lane * 3 + 0];
  float l1 = gh * gw2[lane * 3 + 1];
  float l2 = gh * gw2[lane * 3 + 2];
#pragma unroll
  for (int off = 16; off; off >>= 1) {
    l0 += __shfl_xor(l0, off, 32);
    l1 += __shfl_xor(l1, off, 32);
    l2 += __shfl_xor(l2, off, 32);
  }
  if (lane == 0) {
    l0 += gb2[0]; l1 += gb2[1]; l2 += gb2[2];
    float m = fmaxf(l0, fmaxf(l1, l2));
    float e0 = expf(l0 - m), e1 = expf(l1 - m), e2 = expf(l2 - m);
    float s = e0 + e1 + e2;
    gate[n * 3 + 0] = e0 / s;
    gate[n * 3 + 1] = e1 / s;
    gate[n * 3 + 2] = e2 / s;
  }
}

// ---------------- weight pre-pack: f32 [128,Kout] -> bf16 WMMA B-fragment layout ----------------
// Layout (v16bf units): frag[(ct*4 + q)*32 + lane], lane's 16 values are
// K = q*32 + (i>>3)*16 + (lane>>4)*8 + (i&7), column c = ct*16 + (lane&15); zero-padded for c >= Kout.

__global__ void k_pack_w(const float* __restrict__ W, __bf16* __restrict__ Wp,
                         int Kout, int total) {
  int idx = blockIdx.x * blockDim.x + threadIdx.x;
  if (idx >= total) return;
  int i    = idx & 15;
  int lane = (idx >> 4) & 31;
  int q    = (idx >> 9) & 3;
  int ct   = idx >> 11;
  int k = q * 32 + (i >> 3) * 16 + (lane >> 4) * 8 + (i & 7);
  int c = ct * 16 + (lane & 15);
  Wp[idx] = (c < Kout) ? (__bf16)W[(long)k * Kout + c] : (__bf16)0.f;
}

// ---------------- WMMA bf16 GEMM: C[:,colOff:colOff+Kout] = A@Wp (+ A2@Wp2) + bias ----------------
// A,A2: [N,128] f32 row-major. Wp: pre-packed bf16 fragments (see k_pack_w).
// One wave per 16x16 C tile. blockDim.x = 32 * ceil(Kout/16); grid.x = ceil(N/16).

__device__ __forceinline__ v8f wmma_k128(const float* __restrict__ ap,
                                         const v16bf* __restrict__ bp, v8f acc) {
#pragma unroll
  for (int q = 0; q < 4; ++q) {
    const v4f* p4 = (const v4f*)(ap + q * 32);
    float av[16];
    *(v4f*)(av + 0)  = p4[0];  // K = kA + 0..3
    *(v4f*)(av + 4)  = p4[1];  // K = kA + 4..7
    *(v4f*)(av + 8)  = p4[4];  // K = kA + 16..19
    *(v4f*)(av + 12) = p4[5];  // K = kA + 20..23
    v16bf a;
#pragma unroll
    for (int i = 0; i < 16; ++i) a[i] = (__bf16)av[i];
    v16bf b = bp[q * 32];
    acc = __builtin_amdgcn_wmma_f32_16x16x32_bf16(false, a, false, b, (short)0, acc,
                                                  false, false);
  }
  return acc;
}

__global__ void k_gemm(const float* __restrict__ A, const __bf16* __restrict__ Wp,
                       const float* __restrict__ A2, const __bf16* __restrict__ Wp2,
                       const float* __restrict__ bias, float* __restrict__ C,
                       int N, int Kout, int ldc, int colOff, int relu) {
  int lane = threadIdx.x & 31;
  int ct = threadIdx.x >> 5;          // column tile within block
  int rowBase = blockIdx.x * 16;
  int arow = lane & 15;
  int kA = (lane >> 4) * 8;           // A: lanes 0-15 hold K 0-7 & 16-23; 16-31 hold 8-15 & 24-31
  int gr = rowBase + arow; if (gr > N - 1) gr = N - 1;  // clamp (keep EXEC full for WMMA)

  const float* ap  = A + (long)gr * F + kA;
  const v16bf* bp  = (const v16bf*)Wp + (ct * 4) * 32 + lane;

  v8f acc = {0.f, 0.f, 0.f, 0.f, 0.f, 0.f, 0.f, 0.f};
  acc = wmma_k128(ap, bp, acc);
  if (A2) {
    const float* ap2 = A2 + (long)gr * F + kA;
    const v16bf* bp2 = (const v16bf*)Wp2 + (ct * 4) * 32 + lane;
    acc = wmma_k128(ap2, bp2, acc);
  }

  // C/D layout: lanes 0-15 -> rows 0-7, lanes 16-31 -> rows 8-15; col = lane&15
  int c = ct * 16 + (lane & 15);
  int mBase = (lane >> 4) * 8;
  if (c < Kout) {
    float bv = bias ? bias[c] : 0.f;
#pragma unroll
    for (int r = 0; r < 8; ++r) {
      int rowi = rowBase + mBase + r;
      if (rowi < N) {
        float v = acc[r] + bv;
        if (relu) v = fmaxf(v, 0.f);
        C[(long)rowi * ldc + colOff + c] = v;
      }
    }
  }
}

// ---------------- gated combine + BN stats, BN prep, classifier ----------------

__global__ void k_combine(const float* __restrict__ hg, const float* __restrict__ h2,
                          const float* __restrict__ hs, const float* __restrict__ gate,
                          float* __restrict__ H, float* __restrict__ bnSum,
                          float* __restrict__ bnSq, int N) {
  int f = threadIdx.x;  // 128 threads: one feature each
  float s = 0.f, ss = 0.f;
  for (int n = blockIdx.x; n < N; n += gridDim.x) {
    float g0 = gate[n * 3 + 0], g1 = gate[n * 3 + 1], g2 = gate[n * 3 + 2];
    long i = (long)n * F + f;
    float v = hg[i] * g0 + h2[i] * g1 + hs[i] * g2;
    H[i] = v;
    s += v; ss += v * v;
  }
  atomicAdd(&bnSum[f], s);
  atomicAdd(&bnSq[f], ss);
}

__global__ void k_bnprep(const float* __restrict__ bnSum, const float* __restrict__ bnSq,
                         const float* __restrict__ gamma, const float* __restrict__ beta,
                         float* __restrict__ Ab, float* __restrict__ Bb, int N) {
  int f = threadIdx.x;
  float mu = bnSum[f] / (float)N;
  float var = bnSq[f] / (float)N - mu * mu;
  float a = gamma[f] * rsqrtf(var + 1e-5f);
  Ab[f] = a;
  Bb[f] = beta[f] - mu * a;
}

__global__ void k_cls(const float* __restrict__ H, const float* __restrict__ Ab,
                      const float* __restrict__ Bb, const float* __restrict__ cw,
                      const float* __restrict__ cb, float* __restrict__ out, int N) {
  int n = (blockIdx.x * blockDim.x + threadIdx.x) >> 5;
  int lane = threadIdx.x & 31;
  if (n >= N) return;
  float a0 = 0.f, a1 = 0.f;
#pragma unroll
  for (int i = 0; i < 4; ++i) {
    int f = lane + 32 * i;
    float h = H[(long)n * F + f] * Ab[f] + Bb[f];
    a0 += h * cw[f * 2 + 0];
    a1 += h * cw[f * 2 + 1];
  }
#pragma unroll
  for (int off = 16; off; off >>= 1) {
    a0 += __shfl_xor(a0, off, 32);
    a1 += __shfl_xor(a1, off, 32);
  }
  if (lane == 0) {
    out[(long)n * 2 + 0] = a0 + cb[0];
    out[(long)n * 2 + 1] = a1 + cb[1];
  }
}

// ---------------- launch ----------------

extern "C" void kernel_launch(void* const* d_in, const int* in_sizes, int n_in,
                              void* d_out, int out_size, void* d_ws, size_t ws_size,
                              hipStream_t stream) {
  const float* x    = (const float*)d_in[0];
  const int*   eidx = (const int*)d_in[1];
  const float* fi   = (const float*)d_in[2];
  const float* gcn1_w = (const float*)d_in[3];
  const float* gcn1_b = (const float*)d_in[4];
  const float* gcn2_w = (const float*)d_in[5];
  const float* gcn2_b = (const float*)d_in[6];
  const float* ego_w  = (const float*)d_in[7];
  const float* ego_b  = (const float*)d_in[8];
  const float* hop0_w = (const float*)d_in[9];
  const float* hop0_b = (const float*)d_in[10];
  const float* hop1_w = (const float*)d_in[11];
  const float* hop1_b = (const float*)d_in[12];
  const float* s1wl = (const float*)d_in[13];
  const float* s1wr = (const float*)d_in[14];
  const float* s1b  = (const float*)d_in[15];
  const float* s2wl = (const float*)d_in[16];
  const float* s2wr = (const float*)d_in[17];
  const float* s2b  = (const float*)d_in[18];
  const float* gw1 = (const float*)d_in[19];
  const float* gb1 = (const float*)d_in[20];
  const float* gw2 = (const float*)d_in[21];
  const float* gb2 = (const float*)d_in[22];
  const float* bng = (const float*)d_in[23];
  const float* bnb = (const float*)d_in[24];
  const float* cls_w = (const float*)d_in[25];
  const float* cls_b = (const float*)d_in[26];

  const int N = in_sizes[0] / F;
  const int E = in_sizes[1] / 2;
  const int* row = eidx;
  const int* col = eidx + E;
  const int HOP_DIM = 42, LAST_DIM = F - 2 * 42;  // 44
  const long NF = (long)N * F;

  // workspace carve (256B aligned)
  char* w = (char*)d_ws;
  auto carve = [&](size_t bytes) -> void* {
    void* p = (void*)w;
    w += (bytes + 255) & ~(size_t)255;
    return p;
  };
  float* XW    = (float*)carve(NF * 4);   // x_w; later reused as h_sage output
  float* MEAN  = (float*)carve(NF * 4);   // mean_nbr / h_hop1; later reused as h_sage1
  float* TMP1  = (float*)carve(NF * 4);
  float* TMP2  = (float*)carve(NF * 4);
  float* HGCN  = (float*)carve(NF * 4);
  float* H2CAT = (float*)carve(NF * 4);
  float* degr  = (float*)carve((size_t)N * 4);
  float* degc  = (float*)carve((size_t)N * 4);
  float* dinv  = (float*)carve((size_t)N * 4);
  float* gate  = (float*)carve((size_t)N * 3 * 4);
  float* bnSum = (float*)carve(F * 4);
  float* bnSq  = (float*)carve(F * 4);
  float* bnA   = (float*)carve(F * 4);
  float* bnB   = (float*)carve(F * 4);
  // packed bf16 weights: nct tiles * 2048 bf16 each
  const int P128 = 8 * 2048, P48 = 3 * 2048;
  __bf16* Pgcn1 = (__bf16*)carve(P128 * 2);
  __bf16* Pgcn2 = (__bf16*)carve(P128 * 2);
  __bf16* Pego  = (__bf16*)carve(P48 * 2);
  __bf16* Phop0 = (__bf16*)carve(P48 * 2);
  __bf16* Phop1 = (__bf16*)carve(P48 * 2);
  __bf16* Ps1l  = (__bf16*)carve(P128 * 2);
  __bf16* Ps1r  = (__bf16*)carve(P128 * 2);
  __bf16* Ps2l  = (__bf16*)carve(P128 * 2);
  __bf16* Ps2r  = (__bf16*)carve(P128 * 2);

  const dim3 EW(4096), TB(256);
  const int GT = (N + 15) / 16;       // gemm row tiles
  const int SC = 8192;                 // scatter blocks
  const int NW = (N + 7) / 8;          // wave-per-node grids (256 thr = 8 waves)
  const int PG128 = (P128 + 255) / 256, PG48 = (P48 + 255) / 256;

  // pack all weights to bf16 WMMA fragments
  k_pack_w<<<PG128, 256, 0, stream>>>(gcn1_w, Pgcn1, F, P128);
  k_pack_w<<<PG128, 256, 0, stream>>>(gcn2_w, Pgcn2, F, P128);
  k_pack_w<<<PG48, 256, 0, stream>>>(ego_w,  Pego,  HOP_DIM, P48);
  k_pack_w<<<PG48, 256, 0, stream>>>(hop0_w, Phop0, HOP_DIM, P48);
  k_pack_w<<<PG48, 256, 0, stream>>>(hop1_w, Phop1, LAST_DIM, P48);
  k_pack_w<<<PG128, 256, 0, stream>>>(s1wl, Ps1l, F, P128);
  k_pack_w<<<PG128, 256, 0, stream>>>(s1wr, Ps1r, F, P128);
  k_pack_w<<<PG128, 256, 0, stream>>>(s2wl, Ps2l, F, P128);
  k_pack_w<<<PG128, 256, 0, stream>>>(s2wr, Ps2r, F, P128);

  // degrees, x_w, dinv
  k_zero<<<256, 256, 0, stream>>>(degr, (long)2 * N);   // degr & degc contiguous
  k_zero<<<1, 256, 0, stream>>>(bnSum, 256);            // bnSum & bnSq contiguous
  k_xw<<<EW, TB, 0, stream>>>(x, fi, XW, NF);
  k_deg<<<2048, 256, 0, stream>>>(row, col, degr, degc, E);
  k_dinv<<<(N + 255) / 256, 256, 0, stream>>>(degc, dinv, N);

  // mean neighbor (also H2GCN hop-1), then delta + gate
  k_zero<<<EW, TB, 0, stream>>>(MEAN, NF);
  k_scatter<<<SC, F, 0, stream>>>(XW, col, row, nullptr, MEAN, E);
  k_stats<<<NW, 256, 0, stream>>>(XW, MEAN, degr, gw1, gb1, gw2, gb2, gate, N);

  // ---- GCN path ----
  k_gemm<<<GT, 256, 0, stream>>>(XW, Pgcn1, nullptr, nullptr, nullptr, TMP1, N, F, F, 0, 0);
  k_zero<<<EW, TB, 0, stream>>>(TMP2, NF);
  k_scatter<<<SC, F, 0, stream>>>(TMP1, row, col, dinv, TMP2, E);
  k_gcn_epi<<<EW, TB, 0, stream>>>(TMP2, TMP1, dinv, gcn1_b, TMP2, NF, 1);
  k_gemm<<<GT, 256, 0, stream>>>(TMP2, Pgcn2, nullptr, nullptr, nullptr, TMP1, N, F, F, 0, 0);
  k_zero<<<EW, TB, 0, stream>>>(TMP2, NF);
  k_scatter<<<SC, F, 0, stream>>>(TMP1, row, col, dinv, TMP2, E);
  k_gcn_epi<<<EW, TB, 0, stream>>>(TMP2, TMP1, dinv, gcn2_b, HGCN, NF, 0);

  // ---- H2GCN path (relu applied per segment == relu(concat)) ----
  k_gemm<<<GT, 96, 0, stream>>>(XW, Pego, nullptr, nullptr, ego_b, H2CAT, N, HOP_DIM, F, 0, 1);
  k_gemm<<<GT, 96, 0, stream>>>(MEAN, Phop0, nullptr, nullptr, hop0_b, H2CAT, N, HOP_DIM, F, HOP_DIM, 1);
  k_zero<<<EW, TB, 0, stream>>>(TMP1, NF);
  k_scatter<<<SC, F, 0, stream>>>(MEAN, col, row, nullptr, TMP1, E);
  k_rowdiv<<<EW, TB, 0, stream>>>(TMP1, degr, NF);
  k_gemm<<<GT, 96, 0, stream>>>(TMP1, Phop1, nullptr, nullptr, hop1_b, H2CAT, N, LAST_DIM, F, 2 * HOP_DIM, 1);

  // ---- SAGE path (dual-A fused GEMMs) ----
  k_zero<<<EW, TB, 0, stream>>>(TMP2, NF);
  k_scatter<<<SC, F, 0, stream>>>(XW, row, col, nullptr, TMP2, E);
  k_rowdiv<<<EW, TB, 0, stream>>>(TMP2, degc, NF);
  k_gemm<<<GT, 256, 0, stream>>>(TMP2, Ps1l, XW, Ps1r, s1b, MEAN, N, F, F, 0, 1);  // h_sage1 -> MEAN
  k_zero<<<EW, TB, 0, stream>>>(TMP1, NF);
  k_scatter<<<SC, F, 0, stream>>>(MEAN, row, col, nullptr, TMP1, E);
  k_rowdiv<<<EW, TB, 0, stream>>>(TMP1, degc, NF);
  k_gemm<<<GT, 256, 0, stream>>>(TMP1, Ps2l, MEAN, Ps2r, s2b, XW, N, F, F, 0, 0);  // h_sage -> XW

  // ---- combine + BN + classifier ----
  k_combine<<<1024, F, 0, stream>>>(HGCN, H2CAT, XW, gate, TMP1, bnSum, bnSq, N);
  k_bnprep<<<1, F, 0, stream>>>(bnSum, bnSq, bng, bnb, bnA, bnB, N);
  k_cls<<<NW, 256, 0, stream>>>(TMP1, bnA, bnB, cls_w, cls_b, (float*)d_out, N);

  (void)n_in; (void)out_size; (void)ws_size;
}
